// PointToAnchor_56942676410709
// MI455X (gfx1250) — matrix operand order repaired and checked by reading, not compiled, not run on hardware
//
#include <hip/hip_runtime.h>

// Problem constants (from reference)
#define NN   4
#define CC   256
#define HH   128
#define WW2  128
#define FF   16
#define NBOX 100
#define MM   (NBOX * NN * FF * FF)   // 102400

typedef __attribute__((ext_vector_type(16))) __bf16 v16bf;
typedef __attribute__((ext_vector_type(8)))  __bf16 v8bf;
typedef __attribute__((ext_vector_type(4)))  __bf16 v4bf;
typedef __attribute__((ext_vector_type(8)))  float  v8f;
typedef __attribute__((ext_vector_type(4)))  float  v4f;

static __device__ __forceinline__ v16bf cat88(v8bf lo, v8bf hi) {
  return __builtin_shufflevector(lo, hi, 0,1,2,3,4,5,6,7,8,9,10,11,12,13,14,15);
}
static __device__ __forceinline__ v8bf cat44(v4bf lo, v4bf hi) {
  return __builtin_shufflevector(lo, hi, 0,1,2,3,4,5,6,7);
}

// A fragment (16x32 bf16, lane = row L&15, h = L>>4):
// elements 0..7  <- row[k0 + h*8 .. +7], elements 8..15 <- row[k0 + 16 + h*8 .. +7]
static __device__ __forceinline__ v16bf a_frag_bf16(const __bf16* rowp, int k0, int h) {
  v8bf lo = *(const v8bf*)(rowp + k0 + h * 8);
  v8bf hi = *(const v8bf*)(rowp + k0 + 16 + h * 8);
  return cat88(lo, hi);
}
static __device__ __forceinline__ v16bf a_frag_f32(const float* rowp, int k0, int h) {
  v4f f0 = *(const v4f*)(rowp + k0 + h * 8);
  v4f f1 = *(const v4f*)(rowp + k0 + h * 8 + 4);
  v4f f2 = *(const v4f*)(rowp + k0 + 16 + h * 8);
  v4f f3 = *(const v4f*)(rowp + k0 + 16 + h * 8 + 4);
  v8bf lo = cat44(__builtin_convertvector(f0, v4bf), __builtin_convertvector(f1, v4bf));
  v8bf hi = cat44(__builtin_convertvector(f2, v4bf), __builtin_convertvector(f3, v4bf));
  return cat88(lo, hi);
}
// B fragment (32x16 bf16, lane = col L&15): elements 0..7 <- K k0 + h*8 .. +7, 8..15 <- K k0 + 16 + h*8 .. +7
static __device__ __forceinline__ v16bf b_frag_bf16(const __bf16* colp, int k0, int h) {
  v8bf lo = *(const v8bf*)(colp + k0 + h * 8);
  v8bf hi = *(const v8bf*)(colp + k0 + 16 + h * 8);
  return cat88(lo, hi);
}
#define WMMA_BF16(a, b, c) \
  __builtin_amdgcn_wmma_f32_16x16x32_bf16(false, (a), false, (b), (short)0, (c), false, false)

// ---------------------------------------------------------------------------
// Kernel 0: convert the three weight matrices to bf16 in ws (contiguous).
// ---------------------------------------------------------------------------
__global__ void k0_cvt_weights(const float* __restrict__ Ww, const float* __restrict__ Wf,
                               const float* __restrict__ Wpw, __bf16* __restrict__ dst) {
  int i = blockIdx.x * blockDim.x + threadIdx.x;
  if (i < 65536)        dst[i] = (__bf16)Ww[i];
  else if (i < 131072)  dst[i] = (__bf16)Wf[i - 65536];
  else if (i < 147456)  dst[i] = (__bf16)Wpw[i - 131072];
}

// ---------------------------------------------------------------------------
// Kernel 1: per 16-row tile of x:
//   feat_w = relu(x@Ww^T+bw)   -> LDS (bf16, never hits HBM)
//   feat_f = relu(x@Wf^T+bf)   -> d_out fused region (f32)
//   point_weight = feat_w@Wpw^T+bpw -> ws (bf16, [m][ab] layout)
// One wave per workgroup; 288 WMMAs per wave.
// ---------------------------------------------------------------------------
__global__ __launch_bounds__(32)
void k1_rowtile(const float* __restrict__ x,
                const float* __restrict__ bw, const float* __restrict__ bfv,
                const float* __restrict__ bpw,
                const __bf16* __restrict__ Wwb, const __bf16* __restrict__ Wfb,
                const __bf16* __restrict__ Wpwb,
                float* __restrict__ fused, __bf16* __restrict__ pwb) {
  __shared__ __bf16 featw[16 * CC];   // 8 KB
  const int L = threadIdx.x;
  const int r = L & 15, h = L >> 4;
  const int m0 = blockIdx.x * 16;
  const float* xrow = x + (size_t)(m0 + r) * CC;

  // Cache all A fragments of this x row tile (8 K-tiles).
  v16bf ax[8];
#pragma unroll
  for (int kt = 0; kt < 8; ++kt) ax[kt] = a_frag_f32(xrow, kt * 32, h);

  // GEMM 1: feat_w -> LDS as bf16
  for (int nt = 0; nt < 16; ++nt) {
    const int col = nt * 16 + r;
    const __bf16* brow = Wwb + (size_t)col * CC;
    v8f acc = {};
#pragma unroll
    for (int kt = 0; kt < 8; ++kt) acc = WMMA_BF16(ax[kt], b_frag_bf16(brow, kt * 32, h), acc);
    const float bias = bw[col];
#pragma unroll
    for (int i = 0; i < 8; ++i) {
      float v = acc[i] + bias;
      featw[(i + h * 8) * CC + col] = (__bf16)(v > 0.f ? v : 0.f);
    }
  }
  // GEMM 2: feat_f -> fused region (f32); kernel 2 adds `out` into it.
  for (int nt = 0; nt < 16; ++nt) {
    const int col = nt * 16 + r;
    const __bf16* brow = Wfb + (size_t)col * CC;
    v8f acc = {};
#pragma unroll
    for (int kt = 0; kt < 8; ++kt) acc = WMMA_BF16(ax[kt], b_frag_bf16(brow, kt * 32, h), acc);
    const float bias = bfv[col];
#pragma unroll
    for (int i = 0; i < 8; ++i) {
      float v = acc[i] + bias;
      fused[(size_t)(m0 + i + h * 8) * CC + col] = (v > 0.f ? v : 0.f);
    }
  }
  __syncthreads();

  // GEMM 3: point_weight = feat_w @ Wpw^T + bpw -> ws bf16 (M x 64)
  const __bf16* arow = featw + r * CC;
  v16bf af[8];
#pragma unroll
  for (int kt = 0; kt < 8; ++kt) af[kt] = a_frag_bf16(arow, kt * 32, h);
  for (int nt = 0; nt < 4; ++nt) {
    const int col = nt * 16 + r;
    const __bf16* brow = Wpwb + (size_t)col * CC;
    v8f acc = {};
#pragma unroll
    for (int kt = 0; kt < 8; ++kt) acc = WMMA_BF16(af[kt], b_frag_bf16(brow, kt * 32, h), acc);
    const float bias = bpw[col];
#pragma unroll
    for (int i = 0; i < 8; ++i)
      pwb[(size_t)(m0 + i + h * 8) * 64 + col] = (__bf16)(acc[i] + bias);
  }
}

// ---------------------------------------------------------------------------
// Kernel 2: per (n,p,q) group: out[k,c] = sum_ab pw[k,ab] * feature[n,c,p*8+a,q*8+b]
// fb tile (64x256) staged once in LDS as bf16 [c][ab]; 7 waves = 7 k-tiles.
// Epilogue: out -> d_out out region; fused = feat_f + out (read-modify of fused).
// ---------------------------------------------------------------------------
__global__ __launch_bounds__(224)
void k2_einsum(const float* __restrict__ feat, const __bf16* __restrict__ pwb,
               float* __restrict__ fused, float* __restrict__ outp) {
  __shared__ __bf16 Bs[CC * 64];   // 32 KB, [c][ab]
  const int g = blockIdx.x;
  const int n = g >> 8, p = (g >> 4) & 15, q = g & 15;
  const int tid = threadIdx.x;

  // Cooperative stage: feature[n, c, p*8+a, q*8+b] -> Bs[c*64 + a*8 + b] (bf16)
  const float* fbase = feat + (size_t)n * CC * HH * WW2 + (size_t)(p * 8) * WW2 + q * 8;
  for (int idx = tid; idx < 4096; idx += 224) {
    const int c = idx >> 4, rem = idx & 15;
    const int a = rem >> 1, b0 = (rem & 1) * 4;
    v4f f = *(const v4f*)(fbase + (size_t)c * HH * WW2 + a * WW2 + b0);
    *(v4bf*)&Bs[c * 64 + rem * 4] = __builtin_convertvector(f, v4bf);
  }
  __syncthreads();

  const int wave = tid >> 5, L = tid & 31;
  const int r = L & 15, h = L >> 4;
  const int kbase = wave * 16;
  int krow = kbase + r; if (krow > NBOX - 1) krow = NBOX - 1;   // clamp: EXEC stays all-1s
  const __bf16* arow = pwb + ((size_t)krow * (NN * FF * FF) + n * FF * FF + p * FF + q) * 64;
  const v16bf a0 = a_frag_bf16(arow, 0, h);
  const v16bf a1 = a_frag_bf16(arow, 32, h);

  for (int ct = 0; ct < 16; ++ct) {
    const __bf16* brow = Bs + (ct * 16 + r) * 64;
    v8f acc = {};
    acc = WMMA_BF16(a0, b_frag_bf16(brow, 0, h), acc);
    acc = WMMA_BF16(a1, b_frag_bf16(brow, 32, h), acc);
    const int c = ct * 16 + r;
#pragma unroll
    for (int i = 0; i < 8; ++i) {
      const int kr = kbase + i + h * 8;
      if (kr < NBOX) {                      // divergence only around stores, not WMMA
        const size_t off = ((size_t)kr * (NN * FF * FF) + n * FF * FF + p * FF + q) * CC + c;
        const float o = acc[i];
        outp[off] = o;
        fused[off] = fused[off] + o;        // feat_f (written by k1) + out
      }
    }
  }
}

extern "C" void kernel_launch(void* const* d_in, const int* in_sizes, int n_in,
                              void* d_out, int out_size, void* d_ws, size_t ws_size,
                              hipStream_t stream) {
  const float* feature = (const float*)d_in[0];
  const float* rois    = (const float*)d_in[1];
  const float* Ww      = (const float*)d_in[2];
  const float* bw      = (const float*)d_in[3];
  const float* Wf      = (const float*)d_in[4];
  const float* bfv     = (const float*)d_in[5];
  const float* Wpw     = (const float*)d_in[6];
  const float* bpw     = (const float*)d_in[7];
  (void)in_sizes; (void)n_in; (void)out_size; (void)ws_size;

  // Workspace layout (bf16): Ww (65536) | Wf (65536) | Wpw (16384) | pw (M*64)
  __bf16* Wwb  = (__bf16*)d_ws;
  __bf16* Wfb  = Wwb + 65536;
  __bf16* Wpwb = Wfb + 65536;
  __bf16* pwb  = Wpwb + 16384;               // 13.1 MB

  float* fused = (float*)d_out;              // output 0: fused (M*C)
  float* outp  = fused + (size_t)MM * CC;    // output 1: out   (M*C)

  k0_cvt_weights<<<(147456 + 255) / 256, 256, 0, stream>>>(Ww, Wf, Wpw, Wwb);
  k1_rowtile<<<MM / 16, 32, 0, stream>>>(rois, bw, bfv, bpw, Wwb, Wfb, Wpwb, fused, pwb);
  k2_einsum<<<NN * FF * FF, 224, 0, stream>>>(feature, pwb, fused, outp);
}
